// AllAtomEnergyModel_68745246540386
// MI455X (gfx1250) — compile-verified, matrix operand order checked by reading, not run on hardware
//
#include <hip/hip_runtime.h>
#include <hip/hip_bf16.h>
#include <math.h>

typedef __attribute__((ext_vector_type(16))) _Float16 v16h;
typedef __attribute__((ext_vector_type(8)))  _Float16 v8h;
typedef __attribute__((ext_vector_type(8)))  float    v8f;

#define B_CONST   2
#define N_CONST   1024
#define E_CONST   256
#define H_CONST   256
#define BB_CONST  16     // B * 8 frame replicas
#define P_CONST   128
#define D_CONST   128

// CDNA5 f16 A/B fragment K-index mapping for v_wmma_f32_16x16x32_f16
// (16-bit A-matrix 16x32 layout; B mirrors it with row<->col swapped).
__device__ __forceinline__ int kmap(int e, int half) {
  return ((e >> 3) << 4) | (half << 3) | (e & 7);
}

// Build a 16-half fragment from two contiguous 8-half groups (vector loads).
__device__ __forceinline__ v16h load_frag16(const _Float16* p0, const _Float16* p1) {
  v8h lo = *(const v8h*)p0;
  v8h hi = *(const v8h*)p1;
  return __builtin_shufflevector(lo, hi, 0, 1, 2, 3, 4, 5, 6, 7,
                                 8, 9, 10, 11, 12, 13, 14, 15);
}

// ---------------------------------------------------------------------------
// 1) Per-batch center, covariance, 3x3 symmetric eigendecomposition (Jacobi)
// ---------------------------------------------------------------------------
__global__ void frames_kernel(const float* __restrict__ cords,
                              const float* __restrict__ mask,
                              float* __restrict__ Xc,
                              float* __restrict__ Vout,
                              int Nn) {
  int b   = blockIdx.x;
  int tid = threadIdx.x;
  __shared__ float rbuf[256];
  __shared__ float ctr[3];

  auto reduce = [&](float v) -> float {
    rbuf[tid] = v; __syncthreads();
    for (int s = 128; s > 0; s >>= 1) {
      if (tid < s) rbuf[tid] += rbuf[tid + s];
      __syncthreads();
    }
    float r = rbuf[0]; __syncthreads();
    return r;
  };

  float sm = 0.f, s0 = 0.f, s1 = 0.f, s2 = 0.f;
  for (int n = tid; n < Nn; n += 256) {
    const float* p = cords + ((size_t)(b * Nn + n) * 14 + 1) * 3;
    float m = mask[b * Nn + n];
    sm += m; s0 += p[0] * m; s1 += p[1] * m; s2 += p[2] * m;
  }
  float tm = reduce(sm);
  float t0 = reduce(s0);
  float t1 = reduce(s1);
  float t2 = reduce(s2);
  if (tid == 0) { ctr[0] = t0 / tm; ctr[1] = t1 / tm; ctr[2] = t2 / tm; }
  __syncthreads();

  float cxx = 0.f, cxy = 0.f, cxz = 0.f, cyy = 0.f, cyz = 0.f, czz = 0.f;
  for (int n = tid; n < Nn; n += 256) {
    const float* p = cords + ((size_t)(b * Nn + n) * 14 + 1) * 3;
    float m  = mask[b * Nn + n];
    float x0 = p[0] - ctr[0] * m;
    float x1 = p[1] - ctr[1] * m;
    float x2 = p[2] - ctr[2] * m;
    float* xc = Xc + (size_t)(b * Nn + n) * 3;
    xc[0] = x0; xc[1] = x1; xc[2] = x2;
    cxx += x0 * x0; cxy += x0 * x1; cxz += x0 * x2;
    cyy += x1 * x1; cyz += x1 * x2; czz += x2 * x2;
  }
  cxx = reduce(cxx); cxy = reduce(cxy); cxz = reduce(cxz);
  cyy = reduce(cyy); cyz = reduce(cyz); czz = reduce(czz);

  if (tid == 0) {
    float A[3][3] = {{cxx, cxy, cxz}, {cxy, cyy, cyz}, {cxz, cyz, czz}};
    float V[3][3] = {{1, 0, 0}, {0, 1, 0}, {0, 0, 1}};
    for (int sweep = 0; sweep < 15; ++sweep) {
      for (int pi = 0; pi < 3; ++pi) {
        for (int qi = pi + 1; qi < 3; ++qi) {
          float apq = A[pi][qi];
          if (fabsf(apq) < 1e-18f) continue;
          float theta = (A[qi][qi] - A[pi][pi]) / (2.0f * apq);
          float t = copysignf(1.0f, theta) / (fabsf(theta) + sqrtf(theta * theta + 1.0f));
          float c = rsqrtf(t * t + 1.0f);
          float s = t * c;
          for (int kk = 0; kk < 3; ++kk) {
            float akp = A[kk][pi], akq = A[kk][qi];
            A[kk][pi] = c * akp - s * akq; A[kk][qi] = s * akp + c * akq;
          }
          for (int kk = 0; kk < 3; ++kk) {
            float akp = A[pi][kk], akq = A[qi][kk];
            A[pi][kk] = c * akp - s * akq; A[qi][kk] = s * akp + c * akq;
          }
          for (int kk = 0; kk < 3; ++kk) {
            float vkp = V[kk][pi], vkq = V[kk][qi];
            V[kk][pi] = c * vkp - s * vkq; V[kk][qi] = s * vkp + c * vkq;
          }
        }
      }
    }
    float ev[3]  = {A[0][0], A[1][1], A[2][2]};
    int   ord[3] = {0, 1, 2};
    for (int i = 0; i < 2; ++i)
      for (int j = i + 1; j < 3; ++j)
        if (ev[ord[j]] < ev[ord[i]]) { int t3 = ord[i]; ord[i] = ord[j]; ord[j] = t3; }
    for (int i = 0; i < 3; ++i)
      for (int j = 0; j < 3; ++j)
        Vout[b * 9 + j * 3 + i] = V[j][ord[i]];   // column i = i-th eigenvector
  }
}

// ---------------------------------------------------------------------------
// 2) Build h0 (L, BB, 288) f16: [hX(3) | embedding(256) | zero pad(29)]
// ---------------------------------------------------------------------------
__global__ void build_h0_kernel(const float* __restrict__ Xc,
                                const float* __restrict__ Vm,
                                const int*   __restrict__ seq,
                                const float* __restrict__ emb,
                                _Float16*    __restrict__ out,
                                int Nn, int Kpad) {
  int row = blockIdx.x;           // n * BB + bb
  int n  = row >> 4;
  int bb = row & 15;
  int b  = bb >> 3, o = bb & 7;
  int s = seq[b * Nn + n];
  if (s < 0) s = 82;
  const float* xc = Xc + (size_t)(b * Nn + n) * 3;
  const float* V  = Vm + b * 9;
  _Float16* dst = out + (size_t)row * Kpad;
  for (int c = threadIdx.x; c < Kpad; c += blockDim.x) {
    float val;
    if (c < 3) {
      float dot = xc[0] * V[0 * 3 + c] + xc[1] * V[1 * 3 + c] + xc[2] * V[2 * 3 + c];
      float sign = ((o >> (2 - c)) & 1) ? 1.0f : -1.0f;   // _OPS sign pattern
      val = sign * dot;
    } else if (c < 3 + E_CONST) {
      val = emb[(size_t)s * E_CONST + (c - 3)];
    } else {
      val = 0.0f;
    }
    dst[c] = (_Float16)val;
  }
}

// ---------------------------------------------------------------------------
// 3) Weight f32 -> f16 conversion, packed directly into WMMA B-fragment order:
//    dst[((kt*tilesN + nt)*32 + lane)*16 + e] = src[kt*32 + kmap(e), nt*16 + col]
//    so the GEMM loads each B fragment as one contiguous 32-byte vector.
// ---------------------------------------------------------------------------
__global__ void pack_b_f16(const float* __restrict__ src, _Float16* __restrict__ dst,
                           int rows, int cols, int rowsPad) {
  int idx = blockIdx.x * blockDim.x + threadIdx.x;
  int tilesN = cols >> 4;
  int total = (rowsPad >> 5) * tilesN * 512;
  if (idx >= total) return;
  int e    = idx & 15;
  int lane = (idx >> 4) & 31;
  int tile = idx >> 9;
  int kt = tile / tilesN;
  int nt = tile - kt * tilesN;
  int half = lane >> 4, c = lane & 15;
  int r   = (kt << 5) + kmap(e, half);
  int col = (nt << 4) + c;
  dst[idx] = (_Float16)((r < rows) ? src[(size_t)r * cols + col] : 0.0f);
}

// ---------------------------------------------------------------------------
// 4) WMMA GEMM: C(MxN) = A(MxK) * Bpacked(KxN), f16 in.
//    OUTMODE 0: f16 row-major   1: f32 row-major
//            2: f16 attention-transposed vT[bb][col][l]  (for flash V operand)
//    revL != 0 : A row (l, bb) is read from physical row (L-1-l, bb).
// ---------------------------------------------------------------------------
template <int OUTMODE>
__global__ void gemm_wmma(const _Float16* __restrict__ A,
                          const _Float16* __restrict__ Bp,
                          void* __restrict__ Cout,
                          int M, int N, int K, int lda, int ldc,
                          int revL, int L, int BB) {
  int wave = threadIdx.x >> 5;
  int lane = threadIdx.x & 31;
  int tilesN = N >> 4;
  int idx = blockIdx.x * (blockDim.x >> 5) + wave;
  int mT = idx / tilesN;
  int nT = idx - mT * tilesN;
  if (mT >= (M >> 4)) return;

  int half = lane >> 4;
  int r = lane & 15;
  int c = r;

  int row  = (mT << 4) + r;
  int arow = row;
  if (revL) {
    int l  = row / BB;
    int bb = row - l * BB;
    arow = (L - 1 - l) * BB + bb;
  }
  const _Float16* Ap = A + (size_t)arow * lda;

  v8f acc = {};
  for (int k0 = 0; k0 < K; k0 += 32) {
    v16h af = load_frag16(Ap + k0 + (half << 3), Ap + k0 + 16 + (half << 3));
    const _Float16* bp = Bp + (((size_t)(k0 >> 5) * tilesN + nT) * 32 + lane) * 16;
    v16h bf = *(const v16h*)bp;
    acc = __builtin_amdgcn_wmma_f32_16x16x32_f16(false, af, false, bf,
                                                 (short)0, acc, false, false);
  }

  if (OUTMODE == 1) {
    float* C = (float*)Cout;
#pragma unroll
    for (int i = 0; i < 8; ++i) {
      int orow = (mT << 4) + (half << 3) + i;
      C[(size_t)orow * ldc + (nT << 4) + c] = acc[i];
    }
  } else if (OUTMODE == 0) {
    _Float16* C = (_Float16*)Cout;
#pragma unroll
    for (int i = 0; i < 8; ++i) {
      int orow = (mT << 4) + (half << 3) + i;
      C[(size_t)orow * ldc + (nT << 4) + c] = (_Float16)acc[i];
    }
  } else {
    // vT[(bb*128 + ocol) * L + l] : key-major, contiguous along l per (bb, col)
    _Float16* C = (_Float16*)Cout;
#pragma unroll
    for (int i = 0; i < 8; ++i) {
      int orow = (mT << 4) + (half << 3) + i;
      int l  = orow >> 4;          // BB == 16
      int bb = orow & 15;
      C[((size_t)(bb * 128) + (nT << 4) + c) * L + l] = (_Float16)acc[i];
    }
  }
}

// ---------------------------------------------------------------------------
// 5) Flash attention (one wave per 16 query rows per replica):
//    S = qk^T/sqrt(P) -> online softmax -> O = P*v ; a = q + alpha * O
//    __launch_bounds__(32, 1): single-wave block, lift VGPR cap (no spills).
// ---------------------------------------------------------------------------
__global__ void __launch_bounds__(32, 1)
flash_attn_kernel(const _Float16* __restrict__ q,
                  const _Float16* __restrict__ k,
                  const _Float16* __restrict__ vT,
                  const float* __restrict__ mask,
                  const float* __restrict__ alphaPtr, int dsel,
                  _Float16* __restrict__ aout,
                  int L, int Nn, int rev) {
  const int BB = BB_CONST, P = P_CONST;
  int tilesL = L >> 4;
  int b    = blockIdx.x / tilesL;
  int row0 = (blockIdx.x - b * tilesL) << 4;
  int lane = threadIdx.x;
  int half = lane >> 4;
  int r = lane & 15;
  int c = r;

  __shared__ float    sS[16][33];
  __shared__ _Float16 sP[16][32];
  __shared__ float    sScale[16];
  __shared__ float    rowM[16], rowL[16];
  if (lane < 16) { rowM[lane] = -1e30f; rowL[lane] = 0.0f; }
  __syncthreads();

  // Q fragments are invariant over the key loop: load once.
  const _Float16* qrow = q + ((size_t)(row0 + r) * BB + b) * P;
  v16h qf[4];
#pragma unroll
  for (int kc = 0; kc < 4; ++kc)
    qf[kc] = load_frag16(qrow + kc * 32 + (half << 3),
                         qrow + kc * 32 + 16 + (half << 3));

  // Single per-lane V base; per-group offset is the constant g*16*L.
  const _Float16* vbase = vT + ((size_t)(b * P) + c) * L + (half << 3);

  v8f acc[8] = {};
  const float scale = 0.0883883476483f;   // 1/sqrt(128)

  for (int m0 = 0; m0 < L; m0 += 32) {
    // ---- S tile (16 x 32) = q(16x128) . k^T ----
#pragma unroll
    for (int t = 0; t < 2; ++t) {
      const _Float16* krow = k + ((size_t)(m0 + t * 16 + c) * BB + b) * P;
      v8f s = {};
#pragma unroll
      for (int kc = 0; kc < 4; ++kc) {
        v16h bf = load_frag16(krow + kc * 32 + (half << 3),
                              krow + kc * 32 + 16 + (half << 3));
        s = __builtin_amdgcn_wmma_f32_16x16x32_f16(false, qf[kc], false, bf,
                                                   (short)0, s, false, false);
      }
#pragma unroll
      for (int i = 0; i < 8; ++i)
        sS[(half << 3) + i][t * 16 + c] = s[i] * scale;
    }
    __syncthreads();

    // ---- online softmax over this 32-key chunk (lanes 0..15, one row each) ----
    if (lane < 16) {
      int rr = lane;
      float mOld = rowM[rr], mNew = mOld;
#pragma unroll 4
      for (int j = 0; j < 32; ++j) {
        int mi   = m0 + j;
        int phys = rev ? (L - 1 - mi) : mi;
        float val = sS[rr][j];
        if (mask[(b >> 3) * Nn + phys] == 0.0f) val = -1.0e9f;
        sS[rr][j] = val;
        mNew = fmaxf(mNew, val);
      }
      float corr = __expf(mOld - mNew);
      float ls = rowL[rr] * corr;
#pragma unroll 4
      for (int j = 0; j < 32; ++j) {
        float p = __expf(sS[rr][j] - mNew);
        sP[rr][j] = (_Float16)p;
        ls += p;
      }
      rowM[rr] = mNew; rowL[rr] = ls; sScale[rr] = corr;
    }
    __syncthreads();

    // ---- rescale running O, then O += P(16x32) . v(32x128) ----
#pragma unroll
    for (int i = 0; i < 8; ++i) {
      float f = sScale[(half << 3) + i];
#pragma unroll
      for (int g = 0; g < 8; ++g) acc[g][i] *= f;
    }
    v16h pf = load_frag16(&sP[r][(half << 3)], &sP[r][16 + (half << 3)]);
#pragma unroll
    for (int g = 0; g < 8; ++g) {
      const _Float16* vp = vbase + ((size_t)g << 4) * L + m0;
      v16h bf = load_frag16(vp, vp + 16);
      acc[g] = __builtin_amdgcn_wmma_f32_16x16x32_f16(false, pf, false, bf,
                                                      (short)0, acc[g], false, false);
    }
    __syncthreads();
  }

  if (lane < 16) sScale[lane] = 1.0f / rowL[lane];
  __syncthreads();
  float alpha = alphaPtr[dsel];
#pragma unroll
  for (int i = 0; i < 8; ++i) {
    int rr = (half << 3) + i;
    float inv = sScale[rr];
    size_t base = ((size_t)(row0 + rr) * BB + b) * P;
#pragma unroll
    for (int g = 0; g < 8; ++g) {
      size_t idx = base + (g << 4) + c;
      float qv = (float)q[idx];
      aout[idx] = (_Float16)(qv + alpha * acc[g][i] * inv);
    }
  }
}

// ---------------------------------------------------------------------------
// 6) Sequential SRU recurrence (register state; prefetch next timestep's U).
//    Writes its direction's half into the next layer input, un-reversing d=1.
// ---------------------------------------------------------------------------
__global__ void sru_scan_kernel(const float* __restrict__ U,
                                const float* __restrict__ vpar,
                                const float* __restrict__ bpar,
                                const float* __restrict__ mask,
                                _Float16* __restrict__ hout,
                                int L, int Nn, int Hout, int colOff, int rev) {
  const int BB = BB_CONST, D = D_CONST;
  int t = blockIdx.x * blockDim.x + threadIdx.x;
  if (t >= BB * D) return;
  int bb = t / D;
  int j  = t - bb * D;
  float vf = vpar[j],     vr = vpar[D + j];
  float bf = bpar[j],     br = bpar[D + j];
  float cst = 0.0f;
  const int mrow = (bb >> 3) * Nn;
  for (int l = 0; l < L; ++l) {
    size_t base = ((size_t)l * BB + bb) * 512;
    if (l + 1 < L) __builtin_prefetch(&U[base + (size_t)BB * 512], 0, 1);
    float u0 = U[base + j];
    float uf = U[base + 128 + j];
    float ur = U[base + 256 + j];
    float uh = U[base + 384 + j];
    int phys = rev ? (L - 1 - l) : l;
    float msk = mask[mrow + phys];
    float f  = 1.0f / (1.0f + __expf(-(uf + vf * cst + bf)));
    float cn = f * cst + (1.0f - f) * u0;
    float rg = 1.0f / (1.0f + __expf(-(ur + vr * cst + br)));
    float hv = rg * cn + (1.0f - rg) * uh;
    if (msk == 0.0f) { cn = cst; hv = 0.0f; }
    cst = cn;
    hout[((size_t)phys * BB + bb) * Hout + colOff + j] = (_Float16)hv;
  }
}

// ---------------------------------------------------------------------------
// 7) Frame mean -> SiLU MLP head -> masked squared error (scalar loss)
// ---------------------------------------------------------------------------
__global__ void head_loss_kernel(const _Float16* __restrict__ hfin,
                                 const float* __restrict__ w1, const float* __restrict__ b1,
                                 const float* __restrict__ w2, const float* __restrict__ b2,
                                 const float* __restrict__ hat, const float* __restrict__ mask,
                                 float* __restrict__ loss, int Nn) {
  const int BB = BB_CONST;
  int row = blockIdx.x;          // b * N + n
  int b = row / Nn;
  int n = row - b * Nn;
  int tid = threadIdx.x;
  __shared__ float mh[256];
  __shared__ float t1[256];
  __shared__ float part;

  float s = 0.0f;
#pragma unroll
  for (int o = 0; o < 8; ++o)
    s += (float)hfin[((size_t)(n * BB) + b * 8 + o) * H_CONST + tid];
  mh[tid] = s * 0.125f;
  if (tid == 0) part = 0.0f;
  __syncthreads();

  float acc = b1[tid];
  for (int kk = 0; kk < 256; ++kk) acc += mh[kk] * w1[kk * 256 + tid];
  t1[tid] = acc / (1.0f + __expf(-acc));     // silu
  __syncthreads();

  if (tid < 42) {
    float a2 = b2[tid];
    for (int kk = 0; kk < 256; ++kk) a2 += t1[kk] * w2[kk * 42 + tid];
    float d = a2 - hat[(size_t)row * 42 + tid];
    atomicAdd(&part, d * d * mask[row]);
  }
  __syncthreads();
  if (tid == 0) atomicAdd(loss, part);
}

// ---------------------------------------------------------------------------
// Host orchestration
// ---------------------------------------------------------------------------
extern "C" void kernel_launch(void* const* d_in, const int* in_sizes, int n_in,
                              void* d_out, int out_size, void* d_ws, size_t ws_size,
                              hipStream_t stream) {
  (void)in_sizes; (void)n_in; (void)ws_size;
  const int Nn = N_CONST, L = N_CONST, BB = BB_CONST;
  const int M = L * BB;                        // 16384 rows

  const int*   seq   = (const int*)d_in[0];
  const float* cords = (const float*)d_in[1];
  const float* mask  = (const float*)d_in[2];
  const float* hat   = (const float*)d_in[3];
  const float* emb   = (const float*)d_in[4];
  const float *W1p[3], *Wkp[3], *Wvp[3], *alp[3], *W2p[3], *vpp[3], *bpp[3];
  for (int l = 0; l < 3; ++l) {
    int base = 5 + l * 7;
    W1p[l] = (const float*)d_in[base + 0];
    Wkp[l] = (const float*)d_in[base + 1];
    Wvp[l] = (const float*)d_in[base + 2];
    alp[l] = (const float*)d_in[base + 3];
    W2p[l] = (const float*)d_in[base + 4];
    vpp[l] = (const float*)d_in[base + 5];
    bpp[l] = (const float*)d_in[base + 6];
  }
  const float* wo_w1 = (const float*)d_in[26];
  const float* wo_b1 = (const float*)d_in[27];
  const float* wo_w2 = (const float*)d_in[28];
  const float* wo_b2 = (const float*)d_in[29];

  // ---- workspace carve-up (~78 MB, resident in the 192 MB L2) ----
  char* ws = (char*)d_ws;
  size_t off = 0;
  auto carve = [&](size_t bytes) -> char* {
    char* p = ws + off;
    off += (bytes + 255) & ~(size_t)255;
    return p;
  };
  int Kl[3]  = {288, 256, 256};               // K padded to multiple of 32
  int din[3] = {259, 256, 256};
  float*    Xc  = (float*)carve((size_t)B_CONST * Nn * 3 * 4);
  float*    Vm  = (float*)carve(B_CONST * 9 * 4);
  _Float16* W1h[3]; _Float16* Wkh[3]; _Float16* Wvh[3]; _Float16* W2h[3];
  for (int l = 0; l < 3; ++l) W1h[l] = (_Float16*)carve((size_t)2 * Kl[l] * 128 * 2);
  for (int l = 0; l < 3; ++l) Wkh[l] = (_Float16*)carve((size_t)2 * 128 * 128 * 2);
  for (int l = 0; l < 3; ++l) Wvh[l] = (_Float16*)carve((size_t)2 * 128 * 128 * 2);
  for (int l = 0; l < 3; ++l) W2h[l] = (_Float16*)carve((size_t)2 * 128 * 512 * 2);
  _Float16* buf0 = (_Float16*)carve((size_t)M * 288 * 2);
  _Float16* bufB = (_Float16*)carve((size_t)M * 256 * 2);
  _Float16* bufC = (_Float16*)carve((size_t)M * 256 * 2);
  _Float16* qB   = (_Float16*)carve((size_t)M * 128 * 2);
  _Float16* kB   = (_Float16*)carve((size_t)M * 128 * 2);
  _Float16* vT   = (_Float16*)carve((size_t)M * 128 * 2);  // [bb][p][l]
  _Float16* aB   = (_Float16*)carve((size_t)M * 128 * 2);
  float*    Ubuf = (float*)carve((size_t)M * 512 * 4);

  hipMemsetAsync(d_out, 0, sizeof(float) * (size_t)out_size, stream);

  // ---- frames + h0 ----
  frames_kernel<<<B_CONST, 256, 0, stream>>>(cords, mask, Xc, Vm, Nn);
  build_h0_kernel<<<M, 128, 0, stream>>>(Xc, Vm, seq, emb, buf0, Nn, 288);

  // ---- weight conversion (f32 -> f16, packed into WMMA fragment order) ----
  for (int l = 0; l < 3; ++l) {
    for (int d = 0; d < 2; ++d) {
      int tot = Kl[l] * 128;   // == (Kl/32)*(128/16)*512
      pack_b_f16<<<(tot + 255) / 256, 256, 0, stream>>>(
          W1p[l] + (size_t)d * din[l] * 128, W1h[l] + (size_t)d * Kl[l] * 128,
          din[l], 128, Kl[l]);
      tot = 128 * 128;
      pack_b_f16<<<(tot + 255) / 256, 256, 0, stream>>>(
          Wkp[l] + (size_t)d * 128 * 128, Wkh[l] + (size_t)d * 128 * 128, 128, 128, 128);
      pack_b_f16<<<(tot + 255) / 256, 256, 0, stream>>>(
          Wvp[l] + (size_t)d * 128 * 128, Wvh[l] + (size_t)d * 128 * 128, 128, 128, 128);
      tot = 128 * 512;
      pack_b_f16<<<(tot + 255) / 256, 256, 0, stream>>>(
          W2p[l] + (size_t)d * 128 * 512, W2h[l] + (size_t)d * 128 * 512, 128, 512, 128);
    }
  }

  // ---- 3 bidirectional SRU++ layers ----
  const _Float16* inB[3]  = {buf0, bufB, bufC};
  _Float16*       outB[3] = {bufB, bufC, bufB};
  for (int l = 0; l < 3; ++l) {
    int K = Kl[l];
    for (int d = 0; d < 2; ++d) {
      int tiles  = (M / 16) * (128 / 16);
      int blocks = tiles / 8;
      // q = x_d @ W1[d]   (direction-1 reads reversed rows, zero-copy)
      gemm_wmma<0><<<blocks, 256, 0, stream>>>(inB[l], W1h[l] + (size_t)d * K * 128,
                                               qB, M, 128, K, K, 128, d, L, BB);
      // k = q @ Wk[d]  (row-major)  ;  v = q @ Wv[d]  (attention-transposed)
      gemm_wmma<0><<<blocks, 256, 0, stream>>>(qB, Wkh[l] + (size_t)d * 128 * 128,
                                               kB, M, 128, 128, 128, 128, 0, L, BB);
      gemm_wmma<2><<<blocks, 256, 0, stream>>>(qB, Wvh[l] + (size_t)d * 128 * 128,
                                               vT, M, 128, 128, 128, 128, 0, L, BB);
      // a = q + alpha * softmax(qk^T/sqrt(P)) v   (flash, WMMA)
      flash_attn_kernel<<<BB * (L / 16), 32, 0, stream>>>(qB, kB, vT, mask,
                                                          alp[l], d, aB, L, Nn, d);
      // U = a @ W2[d]  (f32 out for the recurrence)
      tiles  = (M / 16) * (512 / 16);
      blocks = tiles / 8;
      gemm_wmma<1><<<blocks, 256, 0, stream>>>(aB, W2h[l] + (size_t)d * 128 * 512,
                                               Ubuf, M, 512, 128, 128, 512, 0, L, BB);
      // sequential SRU scan; writes this direction's half of the next buffer
      sru_scan_kernel<<<(BB * D_CONST + 255) / 256, 256, 0, stream>>>(
          Ubuf, vpp[l] + (size_t)d * 2 * 128, bpp[l] + (size_t)d * 2 * 128,
          mask, outB[l], L, Nn, 256, d * 128, d);
    }
  }

  // ---- head + loss ----
  head_loss_kernel<<<B_CONST * Nn, 256, 0, stream>>>(
      outB[2], wo_w1, wo_b1, wo_w2, wo_b2, hat, mask, (float*)d_out, Nn);
}
